// InvariantPointAttention_7739531068032
// MI455X (gfx1250) — matrix operand order repaired and checked by reading, not compiled, not run on hardware
//
#include <hip/hip_runtime.h>
#include <math.h>

// ---------------------------------------------------------------------------
// Invariant Point Attention for MI455X (gfx1250, wave32, WMMA 16x16x32 f16)
//
// - All matmul work on v_wmma_f32_16x16x32_f16 (f32 accumulate).
// - Operands laid out so WMMA fragments are 2x16B contiguous per lane:
//   A row-major, B transposed ([n][k]) -> direct global b128 frag loads,
//   no LDS staging except for z16 (kept [j][c] to keep its 151MB write
//   coalesced; o_pair stages it into frag-order LDS, read as ds b128).
// - z (302MB fp32) is read once; the fp16 copy (151MB) fits the 192MB L2,
//   so the o_pair pass re-reads it mostly from L2 (~450MB HBM total).
// ---------------------------------------------------------------------------

typedef __attribute__((ext_vector_type(16))) _Float16 v16h;
typedef __attribute__((ext_vector_type(8)))  _Float16 h8;
typedef __attribute__((ext_vector_type(8)))  float    v8f;
typedef __attribute__((ext_vector_type(4)))  _Float16 h4;

#define NRES 768
#define CN   384
#define CZ   128
#define NH   12
#define CH   16
#define PQ   4
#define PV   8
#define PROJW 1152          // 192 + 384 + 144 + 432
#define CATW  2112          // 192 + 3*96 + 96 + 1536
#define INF_F 100000.0f

// ---- CDNA5 WMMA helpers (layouts per ISA 7.12.2, wave32) -------------------

__device__ inline v8f wmma_f16(v16h a, v16h b, v8f c) {
  return __builtin_amdgcn_wmma_f32_16x16x32_f16(false, a, false, b,
                                                (short)0, c, false, false);
}

// A fragment straight from row-major A (lda in halfs, multiple of 8).
// lane(m,hi): halfs h = k[hi*16+h] -> 16 contiguous halfs of row m.
__device__ inline v16h load_a_direct(const _Float16* A, size_t lda) {
  int lane = threadIdx.x & 31;
  int m = lane & 15, hi = lane >> 4;
  const _Float16* row = A + (size_t)m * lda + hi * 16;
  h8 lo = *(const h8*)(row);
  h8 hh = *(const h8*)(row + 8);
  return __builtin_shufflevector(lo, hh, 0, 1, 2, 3, 4, 5, 6, 7,
                                 8, 9, 10, 11, 12, 13, 14, 15);
}

// B fragment straight from transposed B ([n][k], ldb halfs, multiple of 8).
// lane(n,g): halfs h = B[k=g*16+h][n] -> 16 contiguous halfs of row n.
__device__ inline v16h load_b_direct(const _Float16* BT, size_t ldb) {
  int lane = threadIdx.x & 31;
  int n = lane & 15, g = lane >> 4;
  const _Float16* p = BT + (size_t)n * ldb + g * 16;
  h8 lo = *(const h8*)(p);
  h8 hh = *(const h8*)(p + 8);
  return __builtin_shufflevector(lo, hh, 0, 1, 2, 3, 4, 5, 6, 7,
                                 8, 9, 10, 11, 12, 13, 14, 15);
}

// ---- K0: operand prep ------------------------------------------------------

__global__ void k_prep_s16(const float* s, _Float16* s16) {
  int idx = blockIdx.x * blockDim.x + threadIdx.x;
  if (idx < NRES * CN) s16[idx] = (_Float16)s[idx];
}

// wAllT[col][k] (transposed) = concat of wq|wkv|wqp|wkvp columns.
__global__ void k_prep_w(const float* wq, const float* bq,
                         const float* wkv, const float* bkv,
                         const float* wqp, const float* bqp,
                         const float* wkvp, const float* bkvp,
                         _Float16* wAllT, float* bAll) {
  int idx = blockIdx.x * blockDim.x + threadIdx.x;
  if (idx >= CN * PROJW) return;
  int k = idx / PROJW, col = idx % PROJW;
  float w, b;
  if (col < 192)      { w = wq  [k * 192 + col];         b = bq  [col]; }
  else if (col < 576) { w = wkv [k * 384 + (col - 192)]; b = bkv [col - 192]; }
  else if (col < 720) { w = wqp [k * 144 + (col - 576)]; b = bqp [col - 576]; }
  else                { w = wkvp[k * 432 + (col - 720)]; b = bkvp[col - 720]; }
  wAllT[(size_t)col * CN + k] = (_Float16)w;
  if (k == 0) bAll[col] = b;
}

// wout16T[n][k] = wout[k][n]
__global__ void k_prep_wout(const float* wout, _Float16* wout16T) {
  int idx = blockIdx.x * blockDim.x + threadIdx.x;
  if (idx >= CATW * CN) return;
  int k = idx / CN, n = idx % CN;
  wout16T[(size_t)n * CATW + k] = (_Float16)wout[idx];
}

// ---- K1: P = s @ Wall + b  (768x384 @ 384x1152) ----------------------------
// Block = 1 wave, 16x96 output (6 accumulators share one A frag per K step).

__global__ void k_proj_gemm(const _Float16* s16, const _Float16* wAllT,
                            const float* bAll, float* P) {
  int n0 = blockIdx.x * 96, i0 = blockIdx.y * 16;
  int lane = threadIdx.x;
  v8f acc[6] = {};
  for (int kk = 0; kk < CN; kk += 32) {
    v16h aF = load_a_direct(s16 + (size_t)i0 * CN + kk, CN);
#pragma unroll
    for (int nt = 0; nt < 6; ++nt) {
      v16h bF = load_b_direct(wAllT + (size_t)(n0 + nt * 16) * CN + kk, CN);
      acc[nt] = wmma_f16(aF, bF, acc[nt]);
    }
  }
  int n = lane & 15, hi = lane >> 4;
#pragma unroll
  for (int nt = 0; nt < 6; ++nt)
#pragma unroll
    for (int r = 0; r < 8; ++r) {
      int i = i0 + r + hi * 8, col = n0 + nt * 16 + n;
      P[(size_t)i * PROJW + col] = acc[nt][r] + bAll[col];
    }
}

// ---- K2: rotate points, build augmented Q/K vectors and [v|v_pts]^T --------
// logit = (q*sqrt(1/48))·k + hw*Qp·Kp - 0.5*hw*|Kp|^2  (row-const dropped)

__global__ void k_post_proj(const float* P, const float* rot, const float* trans,
                            const float* head_w,
                            _Float16* aq, _Float16* bk, _Float16* vbT) {
  int i = blockIdx.x, h = threadIdx.x;
  if (h >= NH) return;
  const float* Pi = P + (size_t)i * PROJW;
  const float* R  = rot + i * 9;
  const float* T  = trans + i * 3;
  float hw  = log1pf(__expf(head_w[h])) * 0.13608276348795434f; // sqrt(1/54)
  float shw = sqrtf(hw);
  const float qscale = 0.14433756729740643f;                    // sqrt(1/48)
  float qp[PQ][3], kp[PQ + PV][3];
#pragma unroll
  for (int p = 0; p < PQ; ++p) {
    float raw[3];
    for (int d = 0; d < 3; ++d) raw[d] = Pi[576 + d * 48 + h * PQ + p];
    for (int d = 0; d < 3; ++d)
      qp[p][d] = R[d*3+0]*raw[0] + R[d*3+1]*raw[1] + R[d*3+2]*raw[2] + T[d];
  }
#pragma unroll
  for (int p = 0; p < PQ + PV; ++p) {
    float raw[3];
    for (int d = 0; d < 3; ++d) raw[d] = Pi[720 + d * 144 + h * 12 + p];
    for (int d = 0; d < 3; ++d)
      kp[p][d] = R[d*3+0]*raw[0] + R[d*3+1]*raw[1] + R[d*3+2]*raw[2] + T[d];
  }
  float kn2 = 0.f;
#pragma unroll
  for (int p = 0; p < PQ; ++p)
    for (int d = 0; d < 3; ++d) kn2 += kp[p][d] * kp[p][d];

  _Float16* aqr = aq + ((size_t)h * NRES + i) * 32;
  _Float16* bkr = bk + ((size_t)h * NRES + i) * 32;
  _Float16* vt  = vbT + (size_t)h * 48 * NRES;    // [col][j] per head
#pragma unroll
  for (int c = 0; c < CH; ++c) {
    aqr[c] = (_Float16)(Pi[h * CH + c] * qscale);             // q
    bkr[c] = (_Float16)(Pi[192 + h * 32 + c]);                // k
    vt[(size_t)c * NRES + i] = (_Float16)(Pi[192 + h * 32 + 16 + c]); // v
  }
#pragma unroll
  for (int c = 16; c < 28; ++c) {
    int p = (c - 16) / 3, d = (c - 16) % 3;
    aqr[c] = (_Float16)(shw * qp[p][d]);
    bkr[c] = (_Float16)(shw * kp[p][d]);
  }
  aqr[28] = (_Float16)1.0f;
  bkr[28] = (_Float16)(-0.5f * hw * kn2);
  for (int c = 29; c < 32; ++c) { aqr[c] = (_Float16)0.f; bkr[c] = (_Float16)0.f; }
#pragma unroll
  for (int d = 0; d < 3; ++d)
    for (int p = 0; p < PV; ++p)
      vt[(size_t)(16 + d * 8 + p) * NRES + i] = (_Float16)kp[PQ + p][d];
#pragma unroll
  for (int c = 40; c < 48; ++c) vt[(size_t)c * NRES + i] = (_Float16)0.f;
}

// ---- K3: single pass over z: bbias = sqrt(1/3)*(z@wb+bb); also emit z16 ----

__global__ void k_bias_z(const float* z, const float* wb, const float* bb,
                         float* bbias, _Float16* z16) {
  size_t wave = (size_t)(blockIdx.x * blockDim.x + threadIdx.x) >> 5;
  int lane = threadIdx.x & 31;
  if (wave >= (size_t)NRES * NRES) return;
  const float4 zv = ((const float4*)(z + wave * CZ))[lane];
  h4 zz; zz.x = (_Float16)zv.x; zz.y = (_Float16)zv.y;
  zz.z = (_Float16)zv.z; zz.w = (_Float16)zv.w;
  ((h4*)z16)[wave * 32 + lane] = zz;       // fp16 copy (L2-resident for K6)
  float acc[NH];
#pragma unroll
  for (int h = 0; h < NH; ++h) {
    int c = lane * 4;
    acc[h] = zv.x * wb[(c + 0) * NH + h] + zv.y * wb[(c + 1) * NH + h] +
             zv.z * wb[(c + 2) * NH + h] + zv.w * wb[(c + 3) * NH + h];
  }
#pragma unroll
  for (int s = 16; s >= 1; s >>= 1)
#pragma unroll
    for (int h = 0; h < NH; ++h) acc[h] += __shfl_xor(acc[h], s, 32);
  if (lane == 0) {
#pragma unroll
    for (int h = 0; h < NH; ++h)
      bbias[wave * NH + h] = 0.5773502691896258f * (acc[h] + bb[h]); // sqrt(1/3)
  }
}

// ---- K4: fused logits (1 WMMA per 16x16 tile, K=32 augmented) + softmax ----

__global__ void k_logits(const _Float16* aq, const _Float16* bk,
                         const float* bbias, const float* mask, _Float16* a16) {
  __shared__ float lg[16 * NRES];          // 48 KB
  int h = blockIdx.x / 48, ib = blockIdx.x % 48, i0 = ib * 16;
  int tid = threadIdx.x, lane = tid & 31, w = tid >> 5;

  v16h aF = load_a_direct(aq + ((size_t)h * NRES + i0) * 32, 32);
  for (int jt = w; jt < 48; jt += 4) {     // uniform: 12 iterations per wave
    int j0 = jt * 16;
    v16h bF = load_b_direct(bk + ((size_t)h * NRES + j0) * 32, 32);
    v8f c = {};
    c = wmma_f16(aF, bF, c);
    int n = lane & 15, hi = lane >> 4;
#pragma unroll
    for (int r = 0; r < 8; ++r) {
      int ii = r + hi * 8, j = j0 + n;
      size_t ij = (size_t)(i0 + ii) * NRES + j;
      lg[ii * NRES + j] = c[r] + bbias[ij * NH + h] + INF_F * (mask[ij] - 1.0f);
    }
  }
  __syncthreads();
  int row = tid >> 3, t8 = tid & 7;        // 8 threads per row
  float mx = -3.4e38f;
  for (int j = t8; j < NRES; j += 8) mx = fmaxf(mx, lg[row * NRES + j]);
#pragma unroll
  for (int s = 4; s >= 1; s >>= 1) mx = fmaxf(mx, __shfl_xor(mx, s, 32));
  float sum = 0.f;
  for (int j = t8; j < NRES; j += 8) {
    float e = __expf(lg[row * NRES + j] - mx);
    lg[row * NRES + j] = e; sum += e;
  }
#pragma unroll
  for (int s = 4; s >= 1; s >>= 1) sum += __shfl_xor(sum, s, 32);
  float inv = 1.0f / sum;
  for (int j = t8; j < NRES; j += 8)
    a16[((size_t)h * NRES + i0 + row) * NRES + j] =
        (_Float16)(lg[row * NRES + j] * inv);
}

// ---- K5: per-head  a @ [v | v_pts]  (768x768 @ 768x48), 3 acc per wave -----

__global__ void k_attn_out(const _Float16* a16, const _Float16* vbT, float* ov) {
  int i0 = blockIdx.x * 16, h = blockIdx.y;
  int lane = threadIdx.x;
  const _Float16* Ah = a16 + (size_t)h * NRES * NRES + (size_t)i0 * NRES;
  const _Float16* Vt = vbT + (size_t)h * 48 * NRES;
  v8f acc[3] = {};
  for (int kk = 0; kk < NRES; kk += 32) {
    v16h aF = load_a_direct(Ah + kk, NRES);
#pragma unroll
    for (int nt = 0; nt < 3; ++nt) {
      v16h bF = load_b_direct(Vt + (size_t)(nt * 16) * NRES + kk, NRES);
      acc[nt] = wmma_f16(aF, bF, acc[nt]);
    }
  }
  int n = lane & 15, hi = lane >> 4;
#pragma unroll
  for (int nt = 0; nt < 3; ++nt)
#pragma unroll
    for (int r = 0; r < 8; ++r) {
      int i = i0 + r + hi * 8;
      ov[((size_t)i * NH + h) * 48 + nt * 16 + n] = acc[nt][r];
    }
}

// ---- K6: o_pair[i] = a[:,i,:] @ z16[i]  (16x768 @ 768x128 per residue) -----
// A frags direct (heads >= 12 zeroed); z16 staged into frag-order LDS.

__global__ void k_opair(const _Float16* a16, const _Float16* z16, _Float16* cat16) {
  __shared__ v16h sBf[8 * 32];             // 8 n-tiles, frag order, 8 KB
  _Float16* sB = (_Float16*)sBf;
  int i = blockIdx.x;
  int lane = threadIdx.x;
  int m = lane & 15, hi = lane >> 4;
  v8f acc[8] = {};
  for (int kk = 0; kk < NRES; kk += 32) {
    // A fragment: rows = heads (pad to 16 with zeros), k = j
    v16h aF = {};
    if (m < NH) {
      const _Float16* row = a16 + ((size_t)m * NRES + i) * NRES + kk + hi * 16;
      h8 lo = *(const h8*)(row);
      h8 hh = *(const h8*)(row + 8);
      aF = __builtin_shufflevector(lo, hh, 0, 1, 2, 3, 4, 5, 6, 7,
                                   8, 9, 10, 11, 12, 13, 14, 15);
    }
    // stage B: lane reads one k-row (128 contiguous halfs), scatter frag-order
    const _Float16* zrow = z16 + ((size_t)i * NRES + kk + lane) * CZ;
    if (kk + 32 < NRES)
      __builtin_prefetch(z16 + ((size_t)i * NRES + kk + 32 + lane) * CZ, 0, 0);
    int g = lane >> 4, kh = lane & 15;     // this lane's k row -> (g, h=kh)
#pragma unroll
    for (int u = 0; u < CZ; ++u) {
      int nt = u >> 4, n = u & 15;
      sB[((size_t)nt * 32 + (g * 16 + n)) * 16 + kh] = zrow[u];
    }
    __syncthreads();
#pragma unroll
    for (int nt = 0; nt < 8; ++nt)
      acc[nt] = wmma_f16(aF, sBf[nt * 32 + lane], acc[nt]);
    __syncthreads();
  }
  int n = lane & 15;
#pragma unroll
  for (int nt = 0; nt < 8; ++nt)
#pragma unroll
    for (int r = 0; r < 8; ++r) {
      int mm = r + hi * 8;
      if (mm < NH)
        cat16[(size_t)i * CATW + 576 + mm * CZ + nt * 16 + n] = (_Float16)acc[nt][r];
    }
}

// ---- K7: rotate-back o_pt, norms, assemble concat vector (f16) -------------

__global__ void k_finalize(const float* ov, const float* rot, const float* trans,
                           _Float16* cat16) {
  int i = blockIdx.x, tid = threadIdx.x;
  const float* R = rot + i * 9;
  const float* T = trans + i * 3;
  const float* oi = ov + (size_t)i * NH * 48;
  _Float16* ci = cat16 + (size_t)i * CATW;
  for (int t = tid; t < NH * CH; t += 64) {            // o -> [0,192)
    int h = t >> 4, c = t & 15;
    ci[t] = (_Float16)oi[h * 48 + c];
  }
  for (int t = tid; t < NH * PV; t += 64) {            // points
    int h = t >> 3, p = t & 7;
    float g[3], l[3];
#pragma unroll
    for (int d = 0; d < 3; ++d) g[d] = oi[h * 48 + 16 + d * 8 + p] - T[d];
#pragma unroll
    for (int d = 0; d < 3; ++d)                        // R^T (einsum bnij,bnhpi)
      l[d] = R[0 * 3 + d] * g[0] + R[1 * 3 + d] * g[1] + R[2 * 3 + d] * g[2];
    float nrm = sqrtf(l[0]*l[0] + l[1]*l[1] + l[2]*l[2] + 1e-8f);
    ci[192 + 0 * 96 + t] = (_Float16)l[0];
    ci[192 + 1 * 96 + t] = (_Float16)l[1];
    ci[192 + 2 * 96 + t] = (_Float16)l[2];
    ci[480 + t]          = (_Float16)nrm;
  }
}

// ---- K8: out = cat @ wout + bout  (768x2112 @ 2112x384), 6 acc per wave ----

__global__ void k_final_gemm(const _Float16* cat16, const _Float16* wout16T,
                             const float* bout, float* out) {
  int n0 = blockIdx.x * 96, i0 = blockIdx.y * 16;
  int lane = threadIdx.x;
  v8f acc[6] = {};
  for (int kk = 0; kk < CATW; kk += 32) {
    v16h aF = load_a_direct(cat16 + (size_t)i0 * CATW + kk, CATW);
    if (kk + 32 < CATW)
      __builtin_prefetch(cat16 + (size_t)(i0 + (lane & 15)) * CATW + kk + 32, 0, 0);
#pragma unroll
    for (int nt = 0; nt < 6; ++nt) {
      v16h bF = load_b_direct(wout16T + (size_t)(n0 + nt * 16) * CATW + kk, CATW);
      acc[nt] = wmma_f16(aF, bF, acc[nt]);
    }
  }
  int n = lane & 15, hi = lane >> 4;
#pragma unroll
  for (int nt = 0; nt < 6; ++nt)
#pragma unroll
    for (int r = 0; r < 8; ++r) {
      int i = i0 + r + hi * 8, col = n0 + nt * 16 + n;
      out[(size_t)i * CN + col] = acc[nt][r] + bout[col];
    }
}

// ---------------------------------------------------------------------------

extern "C" void kernel_launch(void* const* d_in, const int* in_sizes, int n_in,
                              void* d_out, int out_size, void* d_ws, size_t ws_size,
                              hipStream_t stream) {
  (void)in_sizes; (void)n_in; (void)out_size; (void)ws_size;
  const float* s     = (const float*)d_in[0];
  const float* z     = (const float*)d_in[1];
  const float* rot   = (const float*)d_in[2];
  const float* trans = (const float*)d_in[3];
  const float* mask  = (const float*)d_in[4];
  const float* wq    = (const float*)d_in[5];
  const float* bq    = (const float*)d_in[6];
  const float* wkv   = (const float*)d_in[7];
  const float* bkv   = (const float*)d_in[8];
  const float* wqp   = (const float*)d_in[9];
  const float* bqp   = (const float*)d_in[10];
  const float* wkvp  = (const float*)d_in[11];
  const float* bkvp  = (const float*)d_in[12];
  const float* wb    = (const float*)d_in[13];
  const float* bb    = (const float*)d_in[14];
  const float* hwts  = (const float*)d_in[15];
  const float* wout  = (const float*)d_in[16];
  const float* bout  = (const float*)d_in[17];
  float* out = (float*)d_out;

  char* base = (char*)d_ws;
  size_t off = 0;
  auto take = [&](size_t bytes) -> void* {
    off = (off + 255) & ~(size_t)255;
    void* p = base + off;
    off += bytes;
    return p;
  };
  // Workspace ~209 MB total (z16 dominates; it is sized to sit in L2).
  _Float16* s16    = (_Float16*)take((size_t)NRES * CN * 2);
  _Float16* wAllT  = (_Float16*)take((size_t)CN * PROJW * 2);
  float*    bAll   = (float*)   take((size_t)PROJW * 4);
  _Float16* wout16T= (_Float16*)take((size_t)CATW * CN * 2);
  float*    P      = (float*)   take((size_t)NRES * PROJW * 4);
  _Float16* aq16   = (_Float16*)take((size_t)NH * NRES * 32 * 2);
  _Float16* bk16   = (_Float16*)take((size_t)NH * NRES * 32 * 2);
  _Float16* vbT16  = (_Float16*)take((size_t)NH * 48 * NRES * 2);
  float*    ovbuf  = (float*)   take((size_t)NRES * NH * 48 * 4);
  _Float16* cat16  = (_Float16*)take((size_t)NRES * CATW * 2);
  _Float16* a16    = (_Float16*)take((size_t)NH * NRES * NRES * 2);
  float*    bbias  = (float*)   take((size_t)NRES * NRES * NH * 4);
  _Float16* z16    = (_Float16*)take((size_t)NRES * NRES * CZ * 2);

  k_prep_s16<<<(NRES * CN + 255) / 256, 256, 0, stream>>>(s, s16);
  k_prep_w<<<(CN * PROJW + 255) / 256, 256, 0, stream>>>(
      wq, bq, wkv, bkv, wqp, bqp, wkvp, bkvp, wAllT, bAll);
  k_prep_wout<<<(CATW * CN + 255) / 256, 256, 0, stream>>>(wout, wout16T);
  k_proj_gemm<<<dim3(PROJW / 96, NRES / 16), 32, 0, stream>>>(s16, wAllT, bAll, P);
  k_post_proj<<<NRES, 32, 0, stream>>>(P, rot, trans, hwts, aq16, bk16, vbT16);
  k_bias_z<<<(NRES * NRES * 32) / 256, 256, 0, stream>>>(z, wb, bb, bbias, z16);
  k_logits<<<NH * (NRES / 16), 128, 0, stream>>>(aq16, bk16, bbias, mask, a16);
  k_attn_out<<<dim3(NRES / 16, NH), 32, 0, stream>>>(a16, vbT16, ovbuf);
  k_opair<<<NRES, 32, 0, stream>>>(a16, z16, cat16);
  k_finalize<<<NRES, 64, 0, stream>>>(ovbuf, rot, trans, cat16);
  k_final_gemm<<<dim3(CN / 96, NRES / 16), 32, 0, stream>>>(cat16, wout16T, bout, out);
}
